// SDFNet_13134009991842
// MI455X (gfx1250) — matrix operand order repaired and checked by reading, not compiled, not run on hardware
//
#include <hip/hip_runtime.h>

// ---------------------------------------------------------------------------
// Fully fused SDFNet for MI455X (gfx1250, wave32).
//
// Math reduction:  h_n[c] = w0[c] * G_c^16(x) + 2^16 * b0[c]
//   where G(y) = y + minpool3(y)  if w0[c] > 0   (erosion field  P)
//         G(y) = y + maxpool3(y)  otherwise      (dilation field Q)
// Head conv3x3(8->32) folds to conv3x3 over {P, Q, in-image indicator}
// => im2col K = 27 (padded to 32).
//
// GEMM orientation: D[oc, pixel] = W1eff[oc, K] x im2col[K, pixel] + b1[oc]
//   A = effective weights (M=oc, built once per wave)
//   B = im2col            (N=pixel, branchless build per 16-pixel group)
//   C = b1 broadcast      (bias folded into the WMMA accumulator)
// With oc in M, the ReLU + w2-dot (1x1 conv 32->1) is within-lane over the
// 8 D VGPRs; a single shfl_xor(16) merges the two oc halves.
//
// Erosion loop is float2-vectorized: all LDS traffic is ds_*_b64 on aligned
// pairs (row stride 66 is even). P/Q share one LDS buffer so the head's
// field select is a pure address offset -> one unconditional ds_load.
// ---------------------------------------------------------------------------

typedef __attribute__((ext_vector_type(16))) __bf16          v16bf;
typedef __attribute__((ext_vector_type(16))) unsigned short  v16us;
typedef __attribute__((ext_vector_type(8)))  float           v8f;

#define TI     32            // interior tile (output pixels per block side)
#define HALO   17            // 16 erosion steps + 1 conv3x3 ring
#define PT     (TI + 2*HALO) // 66 padded tile side
#define PCELLS (PT * PT)     // 4356
#define ROWPAIRS (PT / 2)    // 33 float2 pairs per row
#define NPAIRS (PT * ROWPAIRS)
#define IMG    512
#define NB     16

__device__ __forceinline__ unsigned short f2bf_rne(float f) {
  // round-to-nearest-even f32 -> bf16 (used once per wave for weights)
  unsigned int u = __builtin_bit_cast(unsigned int, f);
  unsigned int r = u + 0x7FFFu + ((u >> 16) & 1u);
  return (unsigned short)(r >> 16);
}

__device__ __forceinline__ unsigned short f2bf_trunc(float f) {
  // truncating f32 -> bf16: 1 VALU op; plenty for an 8-bit-mantissa operand
  return (unsigned short)(__builtin_bit_cast(unsigned int, f) >> 16);
}

// Head stage: conv3x3(fields->32) via WMMA bf16 + ReLU + conv1x1(32->1).
// EDGE=false is the check-free fast path for tiles fully inside the image.
template<bool EDGE>
__device__ __forceinline__ void run_head(
    const float* sF, const float* sW,
    const float* __restrict__ b1, const float* __restrict__ w2, float b2v,
    float* __restrict__ out,
    int tid, int R0, int C0, int tileR, int tileC, int batch)
{
  const int lane = tid & 31;
  const int half = lane >> 4;   // K-half selector per 16-bit WMMA layouts
  const int nloc = lane & 15;   // M(=oc%16) for A, N(=pixel%16) for B/D
  const int wave = tid >> 5;

  // A matrices (16x32, M=oc x K): lane M = nloc,
  // elem i -> K = half*16 + i   (matches B's K order)
  v16us au0, au1;
  #pragma unroll
  for (int i = 0; i < 16; ++i) {
    const int K = half * 16 + i;
    au0[i] = (K < 27) ? f2bf_rne(sW[nloc * 27 + K])        : (unsigned short)0;
    au1[i] = (K < 27) ? f2bf_rne(sW[(nloc + 16) * 27 + K]) : (unsigned short)0;
  }
  const v16bf am0 = __builtin_bit_cast(v16bf, au0);
  const v16bf am1 = __builtin_bit_cast(v16bf, au1);

  // C = b1 broadcast over N; D layout: lane L, VGPR r -> M = r + 8*(L/16)
  v8f c0, c1;
  float w2a[8], w2b[8];
  #pragma unroll
  for (int r = 0; r < 8; ++r) {
    const int m = 8 * half + r;
    c0[r]  = b1[m];       c1[r]  = b1[16 + m];
    w2a[r] = w2[m];       w2b[r] = w2[16 + m];
  }

  for (int g = wave; g < (TI * TI) / 16; g += 8) {
    // B matrix (32x16, K x N=pixel): lane N = nloc, elem i -> K = half*16 + i
    // Branchless build: every lane issues the same unconditional ds_load
    // (addresses are always in-tile); zero-pad / indicator / out-of-image
    // handling is done with value selects only.
    const int p  = g * 16 + nloc;          // this lane's pixel id in tile
    const int lr = p >> 5, lc = p & 31;
    v16us bu;
    #pragma unroll
    for (int i = 0; i < 16; ++i) {
      const int K  = half * 16 + i;
      const int Kc = (K < 27) ? K : 26;        // clamp for address gen only
      const int f  = (Kc >= 18) ? 2 : ((Kc >= 9) ? 1 : 0);
      const int k  = Kc - f * 9;
      const int dy = k / 3, dx = k - dy * 3;
      const int rr = lr + (HALO - 1) + dy;     // padded-tile row of neighbor
      const int cc = lc + (HALO - 1) + dx;
      const int fb = (f == 1) ? 1 : 0;         // P at 0, Q at +PCELLS
      const float ld = sF[fb * PCELLS + rr * PT + cc];
      float t = (f == 2) ? 1.0f : ld;          // indicator field
      if (EDGE) {
        const int gR = R0 + rr, gC = C0 + cc;
        const bool in = (gR >= 0) & (gR < IMG) & (gC >= 0) & (gC < IMG);
        t = in ? t : 0.f;                      // conv3x3 zero padding
      }
      bu[i] = (K < 27) ? f2bf_trunc(t) : (unsigned short)0;
    }
    const v16bf bm = __builtin_bit_cast(v16bf, bu);

    v8f d0 = __builtin_amdgcn_wmma_f32_16x16x32_bf16(
        false, am0, false, bm, (short)0, c0, false, false);
    v8f d1 = __builtin_amdgcn_wmma_f32_16x16x32_bf16(
        false, am1, false, bm, (short)0, c1, false, false);

    // relu + dot with w2 over this lane's 16 oc rows (within-lane)
    float acc = 0.f;
    #pragma unroll
    for (int r = 0; r < 8; ++r) {
      acc += w2a[r] * fmaxf(d0[r], 0.f);
      acc += w2b[r] * fmaxf(d1[r], 0.f);
    }
    // merge the two oc halves (lane L <-> L^16), then lanes 0..15 hold
    // complete sums for 16 consecutive pixels -> coalesced store
    acc += __shfl_xor(acc, 16);
    if (half == 0) {
      const int orow = tileR * TI + lr;
      const int ocol = tileC * TI + lc;
      out[(size_t)batch * IMG * IMG + orow * IMG + ocol] = acc + b2v;
    }
  }
}

__global__ __launch_bounds__(256)
void sdf_fused_kernel(const float* __restrict__ x,
                      const float* __restrict__ w0, const float* __restrict__ b0,
                      const float* __restrict__ w1, const float* __restrict__ b1,
                      const float* __restrict__ w2, const float* __restrict__ b2,
                      float* __restrict__ out)
{
  __shared__ float sF[2 * PCELLS];  // fields: P (min) at 0, Q (max) at PCELLS
  __shared__ float sT[2 * PCELLS];  // separable row-min/max temporaries
  __shared__ float sW[32 * 27];     // effective head weights [oc][field*9+k]

  float* sP  = sF;
  float* sQ  = sF + PCELLS;
  float* sTP = sT;
  float* sTQ = sT + PCELLS;

  const int tid   = threadIdx.x;
  const int tileC = blockIdx.x, tileR = blockIdx.y, batch = blockIdx.z;
  const int R0 = tileR * TI - HALO;
  const int C0 = tileC * TI - HALO;
  const float INF = __builtin_inff();

  // ---- prologue (wave 0): fold w0/b0 into effective 3x3 head weights ----
  if (tid < 32) {
    const int oc = tid;
    float accP[9], accQ[9], accC[9];
    #pragma unroll
    for (int k = 0; k < 9; ++k) { accP[k] = 0.f; accQ[k] = 0.f; accC[k] = 0.f; }
    for (int c = 0; c < 8; ++c) {
      const float a  = w0[c];
      const float Bc = 65536.0f * b0[c];   // 2^16 * b0[c]
      #pragma unroll
      for (int k = 0; k < 9; ++k) {
        const float wv = w1[oc * 72 + c * 9 + k];
        if (a > 0.f) accP[k] += a * wv; else accQ[k] += a * wv;
        accC[k] += Bc * wv;
      }
    }
    #pragma unroll
    for (int k = 0; k < 9; ++k) {
      sW[oc * 27 +      k] = accP[k];
      sW[oc * 27 +  9 + k] = accQ[k];
      sW[oc * 27 + 18 + k] = accC[k];
    }
  }

  // ---- load padded x tile; out-of-image = +inf (P) / -inf (Q) ----
  for (int i = tid; i < PCELLS; i += 256) {
    const int r = i / PT, c = i - r * PT;
    const int gR = R0 + r, gC = C0 + c;
    const bool in = (gR >= 0) & (gR < IMG) & (gC >= 0) & (gC < IMG);
    float xv = 0.f;
    if (in) xv = x[(size_t)batch * IMG * IMG + gR * IMG + gC];
    sP[i] = in ? xv :  INF;
    sQ[i] = in ? xv : -INF;
  }
  __syncthreads();

  // ---- 16 iterations of y += min/max-pool3x3(y), separable, float2 ----
  for (int it = 0; it < NB; ++it) {
    // phase 1: horizontal 3-min/max -> sTP/sTQ (b64 LDS traffic)
    for (int j = tid; j < NPAIRS; j += 256) {
      const int r = j / ROWPAIRS;
      const int c = (j - r * ROWPAIRS) * 2;
      const int base = r * PT;
      const float2 mp = *(const float2*)&sP[base + c];
      const float2 mq = *(const float2*)&sQ[base + c];
      const int cl = (c > 0) ? c - 1 : 0;           // clamp (outer-ring values
      const int cr = (c + 2 < PT) ? c + 2 : PT - 1; //  are halo-discarded)
      const float lp = sP[base + cl], rp = sP[base + cr];
      const float lq = sQ[base + cl], rq = sQ[base + cr];
      float2 op, oq;
      op.x = fminf(fminf(lp, mp.x), mp.y);
      op.y = fminf(fminf(mp.x, mp.y), rp);
      oq.x = fmaxf(fmaxf(lq, mq.x), mq.y);
      oq.y = fmaxf(fmaxf(mq.x, mq.y), rq);
      *(float2*)&sTP[base + c] = op;
      *(float2*)&sTQ[base + c] = oq;
    }
    __syncthreads();
    // phase 2: vertical 3-min/max + accumulate into sP/sQ (b64 LDS traffic)
    for (int j = tid; j < NPAIRS; j += 256) {
      const int r = j / ROWPAIRS;
      const int c = (j - r * ROWPAIRS) * 2;
      const int ru = (r > 0) ? r - 1 : 0;
      const int rd = (r < PT - 1) ? r + 1 : PT - 1;
      const float2 up = *(const float2*)&sTP[ru * PT + c];
      const float2 mp = *(const float2*)&sTP[r  * PT + c];
      const float2 dp = *(const float2*)&sTP[rd * PT + c];
      const float2 uq = *(const float2*)&sTQ[ru * PT + c];
      const float2 mq = *(const float2*)&sTQ[r  * PT + c];
      const float2 dq = *(const float2*)&sTQ[rd * PT + c];
      float2 ap = *(const float2*)&sP[r * PT + c];
      float2 aq = *(const float2*)&sQ[r * PT + c];
      ap.x += fminf(fminf(up.x, mp.x), dp.x);
      ap.y += fminf(fminf(up.y, mp.y), dp.y);
      aq.x += fmaxf(fmaxf(uq.x, mq.x), dq.x);
      aq.y += fmaxf(fmaxf(uq.y, mq.y), dq.y);
      *(float2*)&sP[r * PT + c] = ap;
      *(float2*)&sQ[r * PT + c] = aq;
    }
    __syncthreads();
  }

  // ---- head ----
  const float b2v = b2[0];
  const bool edge = (tileR == 0) | (tileR == (IMG / TI - 1)) |
                    (tileC == 0) | (tileC == (IMG / TI - 1));
  if (edge)
    run_head<true >(sF, sW, b1, w2, b2v, out, tid, R0, C0, tileR, tileC, batch);
  else
    run_head<false>(sF, sW, b1, w2, b2v, out, tid, R0, C0, tileR, tileC, batch);
}

extern "C" void kernel_launch(void* const* d_in, const int* in_sizes, int n_in,
                              void* d_out, int out_size, void* d_ws, size_t ws_size,
                              hipStream_t stream) {
  const float* x  = (const float*)d_in[0];
  const float* w0 = (const float*)d_in[1];
  const float* b0 = (const float*)d_in[2];
  const float* w1 = (const float*)d_in[3];
  const float* b1 = (const float*)d_in[4];
  const float* w2 = (const float*)d_in[5];
  const float* b2 = (const float*)d_in[6];
  float* out = (float*)d_out;

  const int batches = in_sizes[0] / (IMG * IMG);   // 8
  dim3 grid(IMG / TI, IMG / TI, batches);          // 16 x 16 x 8 tiles
  sdf_fused_kernel<<<grid, 256, 0, stream>>>(x, w0, b0, w1, b1, w2, b2, out);
}
